// StackTransformerSquash_70171175682504
// MI455X (gfx1250) — matrix operand order repaired and checked by reading, not compiled, not run on hardware
//
#include <hip/hip_runtime.h>

// ---------------------------------------------------------------------------
// Types for CDNA5 WMMA (wave32): V_WMMA_F32_16X16X32_BF16
// ---------------------------------------------------------------------------
typedef __attribute__((ext_vector_type(16))) __bf16 v16bf;
typedef __attribute__((ext_vector_type(8)))  __bf16 v8bf;
typedef __attribute__((ext_vector_type(8)))  float  v8f;
typedef int v4i __attribute__((vector_size(16)));   // payload type of *_b128 builtins

union FragBF16 { v16bf v; v8bf h[2]; };

#define TM 128      // block tile M
#define TN 64       // block tile N
#define TK 64       // K per staging step = 2 WMMA K-steps
#define LDAS 72     // LDS stride (elems) for A tile (128 x 64, padded)
#define LDBS 72     // LDS stride (elems) for B tile (64 x 64, padded)

// CDNA5 async global->LDS (ASYNCcnt) path, if the toolchain exposes it.
#if defined(__AMDGCN__) && __has_builtin(__builtin_amdgcn_global_load_async_to_lds_b128)
#define HAVE_ASYNC_LDS 1
#else
#define HAVE_ASYNC_LDS 0
#endif

#define GAS __attribute__((address_space(1)))
#define LAS __attribute__((address_space(3)))

__device__ __forceinline__ void async_copy_b128(const __bf16* g, __bf16* l) {
#if HAVE_ASYNC_LDS
    __bf16* gnc = const_cast<__bf16*>(g);
    __builtin_amdgcn_global_load_async_to_lds_b128((GAS v4i*)gnc, (LAS v4i*)l, 0, 0);
#else
    (void)g; (void)l;
#endif
}

__device__ __forceinline__ void wait_async_lds() {
#if HAVE_ASYNC_LDS
#if __has_builtin(__builtin_amdgcn_s_wait_asynccnt)
    __builtin_amdgcn_s_wait_asynccnt(0);
#else
    asm volatile("s_wait_asynccnt 0" ::: "memory");
#endif
#endif
}

// ---------------------------------------------------------------------------
// fp32 -> bf16 weight pre-pack
// ---------------------------------------------------------------------------
__global__ __launch_bounds__(256)
void f32_to_bf16_kernel(const float* __restrict__ in, __bf16* __restrict__ out, long n) {
    long i = (long)blockIdx.x * blockDim.x + threadIdx.x;
    long stride = (long)gridDim.x * blockDim.x;
    for (; i < n; i += stride) out[i] = (__bf16)in[i];
}

// ---------------------------------------------------------------------------
// Generic batched WMMA GEMM:  {C fp32, Cb bf16} = act( alpha*A@B^T + bias [+alibi] )
//   A: M x K  fp32 (A_BF16=0) or bf16 (A_BF16=1), row stride lda,
//      per-(b,h) element offset b*sAb + h*sAh
//   B: bf16; N x K rows (B_TRANS=0) or K x N (B_TRANS=1), row stride ldb
//   C/Cb: optional fp32 / bf16 outputs, row stride ldc
// Block: 256 threads = 8 waves; wave tile 32x32 = 2x2 WMMA tiles; 8 wmma/iter.
// ---------------------------------------------------------------------------
template<bool A_BF16, bool B_TRANS, int ACT, bool ALIBI>
__global__ __launch_bounds__(256)
void gemm_wmma(const void* __restrict__ Aptr, const __bf16* __restrict__ Bb,
               const float* __restrict__ bias,
               float* __restrict__ C, __bf16* __restrict__ Cbf,
               int M, int N, int K, int lda, int ldb, int ldc,
               long sAb, long sAh, long sBb, long sBh, long sCb, long sCh,
               int Hdiv, float alpha)
{
    constexpr bool kUseAsync = HAVE_ASYNC_LDS && (A_BF16 || B_TRANS);

    __shared__ __attribute__((aligned(16))) __bf16 As[TM * LDAS];
    __shared__ __attribute__((aligned(16))) __bf16 Bs[TK * LDBS];

    const int tid  = threadIdx.x;
    const int lane = tid & 31;
    const int wid  = tid >> 5;
    const int wm   = (wid & 3) * 32;   // wave row offset in block tile
    const int wn   = (wid >> 2) * 32;  // wave col offset in block tile

    const int p  = blockIdx.z;
    const int bb = p / Hdiv;
    const int hh = p % Hdiv;

    const int m0 = blockIdx.y * TM;
    const int n0 = blockIdx.x * TN;

    const float*  Af = (const float*)Aptr;
    const __bf16* Ab = (const __bf16*)Aptr;

    const long aOff = (long)bb * sAb + (long)hh * sAh;
    const long bOff = (long)bb * sBb + (long)hh * sBh;
    const long cOff = (long)bb * sCb + (long)hh * sCh;

    // staging index precompute
    const int ar  = tid >> 1;          // 0..127 : A row in tile
    const int ac  = (tid & 1) * 32;    // 0/32   : A col half (elems)
    const int bn  = tid >> 2;          // 0..63  : B n row (N-major source)
    const int bk  = (tid & 3) * 16;    // 0..48  : B k chunk
    const int tk  = tid >> 2;          // 0..63  : B k row (K-major source)
    const int tnb = (tid & 3) * 16;    // 0..48  : B n chunk

    const int l16   = lane & 15;
    const int lhalf = lane >> 4;

    v8f zero = {};
    v8f acc[2][2] = {{zero, zero}, {zero, zero}};

    for (int k0 = 0; k0 < K; k0 += TK) {
        __syncthreads();
        // ---- stage A tile (128 x 64 bf16) ----
        if (A_BF16) {
            const __bf16* src = Ab + aOff + (long)(m0 + ar) * lda + k0 + ac;
            __bf16* dst = &As[ar * LDAS + ac];
            if (kUseAsync) {
                async_copy_b128(src,      dst);
                async_copy_b128(src + 8,  dst + 8);
                async_copy_b128(src + 16, dst + 16);
                async_copy_b128(src + 24, dst + 24);
            } else {
                v8bf u0 = ((const v8bf*)src)[0];
                v8bf u1 = ((const v8bf*)src)[1];
                v8bf u2 = ((const v8bf*)src)[2];
                v8bf u3 = ((const v8bf*)src)[3];
                if (k0 + TK < K) __builtin_prefetch(src + TK, 0, 1);
                ((v8bf*)dst)[0] = u0; ((v8bf*)dst)[1] = u1;
                ((v8bf*)dst)[2] = u2; ((v8bf*)dst)[3] = u3;
            }
        } else {
            const float* src = Af + aOff + (long)(m0 + ar) * lda + k0 + ac;
            float4 f[8];
            #pragma unroll
            for (int i = 0; i < 8; ++i) f[i] = ((const float4*)src)[i];
            if (k0 + TK < K) __builtin_prefetch(src + TK, 0, 1);
            __bf16* d = &As[ar * LDAS + ac];
            #pragma unroll
            for (int i = 0; i < 8; ++i) {
                d[4*i+0] = (__bf16)f[i].x; d[4*i+1] = (__bf16)f[i].y;
                d[4*i+2] = (__bf16)f[i].z; d[4*i+3] = (__bf16)f[i].w;
            }
        }
        // ---- stage B tile into K-major LDS (Bs[k][n], 64 x 64 bf16) ----
        if (!B_TRANS) {
            // source rows are N-major (weight / K-matrix layout) -> transpose
            const __bf16* src = Bb + bOff + (long)(n0 + bn) * ldb + k0 + bk;
            v8bf u0 = ((const v8bf*)src)[0];
            v8bf u1 = ((const v8bf*)src)[1];
            if (k0 + TK < K) __builtin_prefetch(src + TK, 0, 1);
            #pragma unroll
            for (int i = 0; i < 8; ++i) Bs[(bk + i) * LDBS + bn]     = u0[i];
            #pragma unroll
            for (int i = 0; i < 8; ++i) Bs[(bk + 8 + i) * LDBS + bn] = u1[i];
        } else {
            // source already K-major (V matrix) -> raw copy
            const __bf16* src = Bb + bOff + (long)(k0 + tk) * ldb + n0 + tnb;
            __bf16* dst = &Bs[tk * LDBS + tnb];
            if (kUseAsync) {
                async_copy_b128(src,     dst);
                async_copy_b128(src + 8, dst + 8);
            } else {
                v8bf u0 = ((const v8bf*)src)[0];
                v8bf u1 = ((const v8bf*)src)[1];
                if (k0 + TK < K) __builtin_prefetch(src + (long)TK * ldb, 0, 1);
                ((v8bf*)dst)[0] = u0; ((v8bf*)dst)[1] = u1;
            }
        }
        if (kUseAsync) wait_async_lds();
        __syncthreads();

        // ---- 2 WMMA K-steps per staged tile ----
        #pragma unroll
        for (int kk = 0; kk < TK; kk += 32) {
            FragBF16 a[2], b[2];
            #pragma unroll
            for (int mi = 0; mi < 2; ++mi) {
                const __bf16* s = &As[(wm + mi * 16 + l16) * LDAS + kk + 8 * lhalf];
                a[mi].h[0] = *(const v8bf*)s;
                a[mi].h[1] = *(const v8bf*)(s + 16);
            }
            #pragma unroll
            for (int ni = 0; ni < 2; ++ni) {
                const __bf16* s = &Bs[(kk + 8 * lhalf) * LDBS + wn + ni * 16 + l16];
                #pragma unroll
                for (int i = 0; i < 8; ++i) {
                    b[ni].v[i]     = s[i * LDBS];
                    b[ni].v[8 + i] = s[(16 + i) * LDBS];
                }
            }
            #pragma unroll
            for (int mi = 0; mi < 2; ++mi)
                #pragma unroll
                for (int ni = 0; ni < 2; ++ni)
                    acc[mi][ni] = __builtin_amdgcn_wmma_f32_16x16x32_bf16(
                        false, a[mi].v, false, b[ni].v, (short)0, acc[mi][ni], false, false);
        }
    }

    // ---- epilogue: scale, bias, ALiBi, GELU; fp32 and/or bf16 outputs ----
    float slope = 0.f;
    if (ALIBI) slope = exp2f(-(float)(hh + 1));   // H=8: 2^{-8(h+1)/8}
    float*  Cp  = C   ? C   + cOff : nullptr;
    __bf16* Cbp = Cbf ? Cbf + cOff : nullptr;
    #pragma unroll
    for (int mi = 0; mi < 2; ++mi) {
        #pragma unroll
        for (int ni = 0; ni < 2; ++ni) {
            #pragma unroll
            for (int r = 0; r < 8; ++r) {
                int m = m0 + wm + mi * 16 + 8 * lhalf + r;   // C layout: lanes16-31 -> M+8
                int n = n0 + wn + ni * 16 + l16;
                float v = acc[mi][ni][r] * alpha;
                if (bias)  v += bias[n];
                if (ALIBI) v -= slope * fabsf((float)(m - n));
                if (ACT == 1) v = 0.5f * v * (1.0f + erff(v * 0.70710678118654752f));
                if (Cp)  Cp[(long)m * ldc + n]  = v;
                if (Cbp) Cbp[(long)m * ldc + n] = (__bf16)v;
            }
        }
    }
}

// ---------------------------------------------------------------------------
// Row softmax (512 cols), writes bf16 probabilities (feeds P@V WMMA directly)
// ---------------------------------------------------------------------------
__global__ __launch_bounds__(256)
void softmax_rows(const float* __restrict__ S, __bf16* __restrict__ P) {
    __shared__ float red[256];
    const long row = blockIdx.x;
    const int tid = threadIdx.x;
    const float* s = S + row * 512;
    float v0 = s[tid], v1 = s[tid + 256];
    red[tid] = fmaxf(v0, v1);
    __syncthreads();
    for (int off = 128; off > 0; off >>= 1) {
        if (tid < off) red[tid] = fmaxf(red[tid], red[tid + off]);
        __syncthreads();
    }
    float mx = red[0];
    __syncthreads();
    float e0 = __expf(v0 - mx), e1 = __expf(v1 - mx);
    red[tid] = e0 + e1;
    __syncthreads();
    for (int off = 128; off > 0; off >>= 1) {
        if (tid < off) red[tid] += red[tid + off];
        __syncthreads();
    }
    float inv = 1.0f / red[0];
    __bf16* p = P + row * 512;
    p[tid]       = (__bf16)(e0 * inv);
    p[tid + 256] = (__bf16)(e1 * inv);
}

// ---------------------------------------------------------------------------
// Y (fp32) / Yb (bf16) = LayerNorm(x + o) * g + b  (D=1024, one block per row)
// ---------------------------------------------------------------------------
__global__ __launch_bounds__(256)
void add_layernorm(const float* __restrict__ X, const float* __restrict__ O,
                   const float* __restrict__ g, const float* __restrict__ b,
                   float* __restrict__ Y, __bf16* __restrict__ Yb) {
    __shared__ float rsum[256], rsq[256];
    const long row = blockIdx.x;
    const int tid = threadIdx.x;
    const float* x = X + row * 1024;
    const float* o = O + row * 1024;
    float v[4]; float s = 0.f, sq = 0.f;
    #pragma unroll
    for (int i = 0; i < 4; ++i) {
        v[i] = x[tid + 256 * i] + o[tid + 256 * i];
        s += v[i]; sq += v[i] * v[i];
    }
    rsum[tid] = s; rsq[tid] = sq;
    __syncthreads();
    for (int off = 128; off > 0; off >>= 1) {
        if (tid < off) { rsum[tid] += rsum[tid + off]; rsq[tid] += rsq[tid + off]; }
        __syncthreads();
    }
    float mean = rsum[0] * (1.0f / 1024.0f);
    float var  = rsq[0] * (1.0f / 1024.0f) - mean * mean;
    float rstd = rsqrtf(var + 1e-5f);
    float*  y  = Y  + row * 1024;
    __bf16* yb = Yb + row * 1024;
    #pragma unroll
    for (int i = 0; i < 4; ++i) {
        int c = tid + 256 * i;
        float r = (v[i] - mean) * rstd * g[c] + b[c];
        y[c]  = r;
        yb[c] = (__bf16)r;
    }
}

// ---------------------------------------------------------------------------
// Orchestration
// ---------------------------------------------------------------------------
extern "C" void kernel_launch(void* const* d_in, const int* in_sizes, int n_in,
                              void* d_out, int out_size, void* d_ws, size_t ws_size,
                              hipStream_t stream) {
    (void)in_sizes; (void)n_in; (void)out_size; (void)ws_size;
    const float* x      = (const float*)d_in[0];
    const float* pre_w  = (const float*)d_in[1];
    const float* pre_b  = (const float*)d_in[2];
    const float* in_w   = (const float*)d_in[3];
    const float* in_b   = (const float*)d_in[4];
    const float* out_w  = (const float*)d_in[5];
    const float* out_b  = (const float*)d_in[6];
    const float* ln1_g  = (const float*)d_in[7];
    const float* ln1_b  = (const float*)d_in[8];
    const float* ff1_w  = (const float*)d_in[9];
    const float* ff1_b  = (const float*)d_in[10];
    const float* ff2_w  = (const float*)d_in[11];
    const float* ff2_b  = (const float*)d_in[12];
    const float* ln2_g  = (const float*)d_in[13];
    const float* ln2_b  = (const float*)d_in[14];
    const float* post_w = (const float*)d_in[15];
    const float* post_b = (const float*)d_in[16];
    float* out = (float*)d_out;

    const long M = 4096;          // B * Tl = 8 * 512
    const int  NH = 8;

    // workspace carve-up
    __bf16* wPre  = (__bf16*)d_ws;
    __bf16* wIn   = wPre  + 1024L * 512;
    __bf16* wOut  = wIn   + 6L * 3072 * 1024;
    __bf16* wF1   = wOut  + 6L * 1024 * 1024;
    __bf16* wF2   = wF1   + 6L * 1024 * 1024;
    __bf16* wPost = wF2   + 6L * 1024 * 1024;
    float*  X     = (float*)(wPost + 1024L * 1024); // fp32 residual stream
    float*  TMP   = X   + M * 1024;                 // fp32 sublayer output
    float*  S     = TMP + M * 1024;                 // 64 x 512 x 512 scores
    __bf16* Xb    = (__bf16*)(S + 64L * 512 * 512); // bf16 residual stream
    __bf16* QKVb  = Xb   + M * 1024;                // 4096 x 3072 bf16
    __bf16* ATTb  = QKVb + M * 3072;                // 4096 x 1024 bf16
    __bf16* FFb   = ATTb + M * 1024;                // 4096 x 1024 bf16
    __bf16* P     = FFb  + M * 1024;                // 64 x 512 x 512 bf16

    dim3 blk(256);

    // ---- weight pre-pack to bf16 ----
    f32_to_bf16_kernel<<<dim3(2048), blk, 0, stream>>>(pre_w,  wPre,  1024L * 512);
    f32_to_bf16_kernel<<<dim3(4096), blk, 0, stream>>>(in_w,   wIn,   6L * 3072 * 1024);
    f32_to_bf16_kernel<<<dim3(4096), blk, 0, stream>>>(out_w,  wOut,  6L * 1024 * 1024);
    f32_to_bf16_kernel<<<dim3(4096), blk, 0, stream>>>(ff1_w,  wF1,   6L * 1024 * 1024);
    f32_to_bf16_kernel<<<dim3(4096), blk, 0, stream>>>(ff2_w,  wF2,   6L * 1024 * 1024);
    f32_to_bf16_kernel<<<dim3(2048), blk, 0, stream>>>(post_w, wPost, 1024L * 1024);

    // ---- pre-projection: X/Xb = reshape(x) @ pre_w^T + pre_b ----
    gemm_wmma<false, false, 0, false><<<dim3(1024 / TN, 4096 / TM, 1), blk, 0, stream>>>(
        x, wPre, pre_b, X, Xb, 4096, 1024, 512, 512, 512, 1024,
        0, 0, 0, 0, 0, 0, 1, 1.0f);

    for (int l = 0; l < 6; ++l) {
        // QKVb = Xb @ in_w^T + in_b   (bf16 out only)
        gemm_wmma<true, false, 0, false><<<dim3(3072 / TN, 4096 / TM, 1), blk, 0, stream>>>(
            Xb, wIn + (long)l * 3072 * 1024, in_b + l * 3072, nullptr, QKVb,
            4096, 3072, 1024, 1024, 1024, 3072, 0, 0, 0, 0, 0, 0, 1, 1.0f);

        // S[b,h] = Q @ K^T / sqrt(128) - slope_h * |i-j|   (64 (b,h) pairs)
        gemm_wmma<true, false, 0, true><<<dim3(512 / TN, 512 / TM, 64), blk, 0, stream>>>(
            QKVb, QKVb + 1024, nullptr, S, nullptr, 512, 512, 128, 3072, 3072, 512,
            512L * 3072, 128, 512L * 3072, 128, 8L * 512 * 512, 512L * 512,
            NH, 0.08838834764831845f);

        // P = softmax(S)  (bf16)
        softmax_rows<<<dim3(64 * 512), blk, 0, stream>>>(S, P);

        // ATTb[b, :, h*128:(h+1)*128] = P @ V   (V consumed K-major, bf16)
        gemm_wmma<true, true, 0, false><<<dim3(128 / TN, 512 / TM, 64), blk, 0, stream>>>(
            P, QKVb + 2048, nullptr, nullptr, ATTb, 512, 128, 512, 512, 3072, 1024,
            8L * 512 * 512, 512L * 512, 512L * 3072, 128, 512L * 1024, 128,
            NH, 1.0f);

        // out-proj: TMP = ATTb @ out_w^T + out_b (fp32 for LN residual)
        gemm_wmma<true, false, 0, false><<<dim3(1024 / TN, 4096 / TM, 1), blk, 0, stream>>>(
            ATTb, wOut + (long)l * 1024 * 1024, out_b + l * 1024, TMP, nullptr,
            4096, 1024, 1024, 1024, 1024, 1024, 0, 0, 0, 0, 0, 0, 1, 1.0f);

        // X/Xb = LN1(X + attn_out)
        add_layernorm<<<dim3(4096), blk, 0, stream>>>(X, TMP, ln1_g + l * 1024, ln1_b + l * 1024, X, Xb);

        // FFb = gelu(Xb @ ff1^T + b)  (bf16)
        gemm_wmma<true, false, 1, false><<<dim3(1024 / TN, 4096 / TM, 1), blk, 0, stream>>>(
            Xb, wF1 + (long)l * 1024 * 1024, ff1_b + l * 1024, nullptr, FFb,
            4096, 1024, 1024, 1024, 1024, 1024, 0, 0, 0, 0, 0, 0, 1, 1.0f);

        // TMP = FFb @ ff2^T + b
        gemm_wmma<true, false, 0, false><<<dim3(1024 / TN, 4096 / TM, 1), blk, 0, stream>>>(
            FFb, wF2 + (long)l * 1024 * 1024, ff2_b + l * 1024, TMP, nullptr,
            4096, 1024, 1024, 1024, 1024, 1024, 0, 0, 0, 0, 0, 0, 1, 1.0f);

        // X/Xb = LN2(X + ff_out)
        add_layernorm<<<dim3(4096), blk, 0, stream>>>(X, TMP, ln2_g + l * 1024, ln2_b + l * 1024, X, Xb);
    }

    // ---- post-projection into d_out ----
    gemm_wmma<true, false, 0, false><<<dim3(1024 / TN, 4096 / TM, 1), blk, 0, stream>>>(
        Xb, wPost, post_b, out, nullptr, 4096, 1024, 1024, 1024, 1024, 1024,
        0, 0, 0, 0, 0, 0, 1, 1.0f);
}
